// CTRNN_64244120813869
// MI455X (gfx1250) — matrix operand order repaired and checked by reading, not compiled, not run on hardware
//
#include <hip/hip_runtime.h>
#include <hip/hip_bf16.h>
#include <math.h>

// ---------------------------------------------------------------------------
// CTRNN on MI455X (gfx1250, wave32, WMMA).
//
//   CVT:   X, W0, W1, Wr -> f16 copies (one-time; weights tiny, X 32MB)
//   GEMM1: U0 = Xh @ W0h^T + b0          [32768,512]x[512,1024]
//   SCAN0: elementwise leaky-integration scan -> H0h (f16, direct)
//   GEMM2: U1 = H0h @ W1h^T + b1         [32768,1024]x[1024,1024]
//   SCAN1: scan -> H1 f32 (d_out hidden region) + H1h f16 (ws)
//   GEMM3: OUT = H1h @ Wrh^T + br        [32768,1024]x[1024,256]
//
// GEMM: v_wmma_f32_16x16x32_f16, f32 accumulate. Wave tile 64x64 = 4x4 WMMA,
// fragments loaded as raw b128 from pre-converted f16 (no inner-loop cvt).
// 192MB L2 holds all f16 operands, so no LDS staging; straight global loads.
// Workspace: ~260 MB.
// ---------------------------------------------------------------------------

typedef __attribute__((ext_vector_type(16))) _Float16 v16h;
typedef __attribute__((ext_vector_type(8)))  _Float16 v8h;
typedef __attribute__((ext_vector_type(4)))  _Float16 v4h;
typedef __attribute__((ext_vector_type(8)))  float    v8f;
typedef __attribute__((ext_vector_type(4)))  float    v4f;

#define T_STEPS 256
#define BATCH   128
#define D_IN    512
#define HID     1024
#define D_OUT   256
#define TB      (T_STEPS * BATCH)   // 32768 rows

// --- A fragment: 16x32 (MxK) f16 from row-major f16 A[lda=K] --------------
// Lane L<16 : row M=L,    K-runs [k+0,8) and [k+16,24)
// Lane L>=16: row M=L-16, K-runs [k+8,16) and [k+24,32)
__device__ __forceinline__ v16h load_a_frag(const _Float16* __restrict__ A,
                                            int lda, int mTile, int k, int lane) {
    const int m   = mTile + (lane & 15);
    const int off = (lane >> 4) * 8;
    const _Float16* p = A + (size_t)m * lda + k + off;
    v8h lo = *reinterpret_cast<const v8h*>(p);        // one b128
    v8h hi = *reinterpret_cast<const v8h*>(p + 16);   // one b128
    return __builtin_shufflevector(lo, hi, 0,1,2,3,4,5,6,7,8,9,10,11,12,13,14,15);
}

// --- B fragment: 32x16 (KxN) f16 from row-major f16 W[N,K] (ldw=K) --------
// Lane L<16 : col N=L,    K-runs [k+0,8) and [k+16,24)
// Lane L>=16: col N=L-16, K-runs [k+8,16) and [k+24,32)
__device__ __forceinline__ v16h load_b_frag(const _Float16* __restrict__ W,
                                            int ldw, int nTile, int k, int lane) {
    const int n   = nTile + (lane & 15);
    const int off = (lane >> 4) * 8;
    const _Float16* p = W + (size_t)n * ldw + k + off;
    v8h lo = *reinterpret_cast<const v8h*>(p);        // one b128
    v8h hi = *reinterpret_cast<const v8h*>(p + 16);   // one b128
    return __builtin_shufflevector(lo, hi, 0,1,2,3,4,5,6,7,8,9,10,11,12,13,14,15);
}

// --- GEMM: C[M,N] = A[M,K] @ W[N,K]^T + bias ------------------------------
// Block: 256 threads = 8 waves (2 in M x 4 in N). Wave tile 64x64 = 4x4 WMMA.
// Block tile: 128 (M) x 256 (N). Requires M%128==0, N%256==0, K%32==0.
__global__ void __launch_bounds__(256)
ctrnn_gemm_wmma(float* __restrict__ C,
                const _Float16* __restrict__ A,
                const _Float16* __restrict__ W,
                const float* __restrict__ bias,
                int M, int N, int K) {
    const int lane  = threadIdx.x & 31;
    const int wave  = threadIdx.x >> 5;
    const int waveM = wave >> 2;        // 0..1
    const int waveN = wave & 3;         // 0..3

    const int mW = blockIdx.x * 128 + waveM * 64;
    const int nW = blockIdx.y * 256 + waveN * 64;

    v8f acc[16];
    #pragma unroll
    for (int i = 0; i < 16; ++i) acc[i] = (v8f){};

    for (int k = 0; k < K; k += 32) {
        // Speculative near-cache prefetch of next K-slab (lanes cover all rows)
        __builtin_prefetch(A + (size_t)(mW + lane)      * K + k + 32, 0, 3);
        __builtin_prefetch(A + (size_t)(mW + 32 + lane) * K + k + 32, 0, 3);
        __builtin_prefetch(W + (size_t)(nW + lane)      * K + k + 32, 0, 3);
        __builtin_prefetch(W + (size_t)(nW + 32 + lane) * K + k + 32, 0, 3);

        v16h a[4], b[4];
        #pragma unroll
        for (int i = 0; i < 4; ++i) a[i] = load_a_frag(A, K, mW + 16 * i, k, lane);
        #pragma unroll
        for (int j = 0; j < 4; ++j) b[j] = load_b_frag(W, K, nW + 16 * j, k, lane);

        #pragma unroll
        for (int i = 0; i < 4; ++i)
            #pragma unroll
            for (int j = 0; j < 4; ++j)
                acc[i * 4 + j] = __builtin_amdgcn_wmma_f32_16x16x32_f16(
                    false, a[i], false, b[j], (short)0, acc[i * 4 + j], false, false);
    }

    // D layout: VGPR r -> row m = r + 8*(lane>=16), col n = lane&15
    const int lh = lane >> 4;
    const int lc = lane & 15;
    #pragma unroll
    for (int j = 0; j < 4; ++j) {
        const float bn = bias[nW + 16 * j + lc];
        #pragma unroll
        for (int i = 0; i < 4; ++i) {
            #pragma unroll
            for (int r = 0; r < 8; ++r) {
                const size_t row = (size_t)(mW + 16 * i + r + 8 * lh) * N;
                C[row + nW + 16 * j + lc] = acc[i * 4 + j][r] + bn;
            }
        }
    }
}

// --- f32 -> f16 conversion (4 elems/thread) -------------------------------
__global__ void __launch_bounds__(256)
cvt_f32_f16(_Float16* __restrict__ dst, const float* __restrict__ src, int n4) {
    const int i = blockIdx.x * blockDim.x + threadIdx.x;
    if (i >= n4) return;
    v4f x = reinterpret_cast<const v4f*>(src)[i];
    v4h y;
    y[0] = (_Float16)x[0]; y[1] = (_Float16)x[1];
    y[2] = (_Float16)x[2]; y[3] = (_Float16)x[3];
    reinterpret_cast<v4h*>(dst)[i] = y;
}

// --- Elementwise leaky-integration scan over T, f16 output ----------------
__global__ void __launch_bounds__(256)
ctrnn_scan_h16(_Float16* __restrict__ Hh,
               const float* __restrict__ U,
               const float* __restrict__ mn,
               const float* __restrict__ an,
               int T, int BH) {
    const int idx = blockIdx.x * blockDim.x + threadIdx.x;
    if (idx >= BH) return;
    float v = 0.0f;
    for (int t = 0; t < T; ++t) {
        const size_t o = (size_t)t * BH + idx;
        v = fmaf(0.9f, v, fmaf(0.1f, U[o], 0.01f * mn[o]));
        Hh[o] = (_Float16)(tanhf(v) + 0.01f * an[o]);
    }
}

// --- Scan with dual output: f32 (final hidden_states) + f16 (GEMM3 input) -
__global__ void __launch_bounds__(256)
ctrnn_scan_dual(float* __restrict__ Hf, _Float16* __restrict__ Hh,
                const float* __restrict__ U,
                const float* __restrict__ mn,
                const float* __restrict__ an,
                int T, int BH) {
    const int idx = blockIdx.x * blockDim.x + threadIdx.x;
    if (idx >= BH) return;
    float v = 0.0f;
    for (int t = 0; t < T; ++t) {
        const size_t o = (size_t)t * BH + idx;
        v = fmaf(0.9f, v, fmaf(0.1f, U[o], 0.01f * mn[o]));
        const float h = tanhf(v) + 0.01f * an[o];
        Hf[o] = h;
        Hh[o] = (_Float16)h;
    }
}

extern "C" void kernel_launch(void* const* d_in, const int* in_sizes, int n_in,
                              void* d_out, int out_size, void* d_ws, size_t ws_size,
                              hipStream_t stream) {
    const float* X   = (const float*)d_in[0];   // [T,B,512]
    const float* W0  = (const float*)d_in[1];   // [1024,512]
    const float* b0  = (const float*)d_in[2];
    const float* W1  = (const float*)d_in[3];   // [1024,1024]
    const float* b1  = (const float*)d_in[4];
    const float* Wr  = (const float*)d_in[5];   // [256,1024]
    const float* br  = (const float*)d_in[6];
    const float* mn0 = (const float*)d_in[7];   // [T,B,1024]
    const float* an0 = (const float*)d_in[8];
    const float* mn1 = (const float*)d_in[9];
    const float* an1 = (const float*)d_in[10];

    float* out = (float*)d_out;                 // outputs [T,B,256] first
    float* H1f = out + (size_t)TB * D_OUT;      // hidden  [T,B,1024]

    // Workspace carve (256B-aligned regions):
    //  U (f32, 128MB) | W0h|W1h|Wrh (f16, 3.5MB) | regionA (64MB: Xh then H1h)
    //  | regionB (64MB: H0h)
    char* w = (char*)d_ws;
    float* U = (float*)w;
    size_t off = (size_t)TB * HID * sizeof(float);
    _Float16* W0h = (_Float16*)(w + off); off += (size_t)HID * D_IN * 2;
    _Float16* W1h = (_Float16*)(w + off); off += (size_t)HID * HID * 2;
    _Float16* Wrh = (_Float16*)(w + off); off += (size_t)D_OUT * HID * 2;
    off = (off + 255) & ~(size_t)255;
    _Float16* Xh  = (_Float16*)(w + off);                 // TB*D_IN halves
    _Float16* H1h = (_Float16*)(w + off);                 // reuses Xh (dead after GEMM1)
    off += (size_t)TB * HID * 2;
    _Float16* H0h = (_Float16*)(w + off);                 // TB*HID halves

    const dim3 blk(256);
    const int  BH = BATCH * HID;

    // One-time f32->f16 conversions
    cvt_f32_f16<<<dim3((TB * D_IN / 4) / 256), blk, 0, stream>>>(Xh,  X,  TB * D_IN / 4);
    cvt_f32_f16<<<dim3((HID * D_IN / 4) / 256), blk, 0, stream>>>(W0h, W0, HID * D_IN / 4);
    cvt_f32_f16<<<dim3((HID * HID / 4) / 256), blk, 0, stream>>>(W1h, W1, HID * HID / 4);
    cvt_f32_f16<<<dim3((D_OUT * HID / 4) / 256), blk, 0, stream>>>(Wrh, Wr, D_OUT * HID / 4);

    // GEMM1: U0 = Xh @ W0h^T + b0
    ctrnn_gemm_wmma<<<dim3(TB / 128, HID / 256), blk, 0, stream>>>(U, Xh, W0h, b0, TB, HID, D_IN);
    // SCAN layer 0 -> H0h (f16)
    ctrnn_scan_h16<<<dim3(BH / 256), blk, 0, stream>>>(H0h, U, mn0, an0, T_STEPS, BH);
    // GEMM2: U1 = H0h @ W1h^T + b1   (reuses U; H1h region still holds dead Xh)
    ctrnn_gemm_wmma<<<dim3(TB / 128, HID / 256), blk, 0, stream>>>(U, H0h, W1h, b1, TB, HID, HID);
    // SCAN layer 1 -> H1 f32 (output) + H1h f16 (GEMM3 input)
    ctrnn_scan_dual<<<dim3(BH / 256), blk, 0, stream>>>(H1f, H1h, U, mn1, an1, T_STEPS, BH);
    // GEMM3: OUT = H1h @ Wrh^T + br
    ctrnn_gemm_wmma<<<dim3(TB / 128, D_OUT / 256), blk, 0, stream>>>(out, H1h, Wrh, br, TB, D_OUT, HID);
}
